// GATRanker_50998441673155
// MI455X (gfx1250) — compile-verified, matrix operand order checked by reading, not run on hardware
//
#include <hip/hip_runtime.h>
#include <hip/hip_bf16.h>

// ---------------- CDNA5 WMMA types ----------------
typedef __attribute__((ext_vector_type(16))) __bf16 v16bf;
typedef __attribute__((ext_vector_type(8)))  float  v8f;

union FragBF { v16bf v; uint4 q[2]; };

// order-preserving float<->uint mapping (for atomic max on floats incl. negatives)
__device__ __forceinline__ unsigned f2o(float f) {
    unsigned b = __float_as_uint(f);
    return (b & 0x80000000u) ? ~b : (b | 0x80000000u);
}
__device__ __forceinline__ float o2f(unsigned u) {
    unsigned b = (u & 0x80000000u) ? (u & 0x7FFFFFFFu) : ~u;
    return __uint_as_float(b);
}

// ---------------- utility kernels ----------------
__global__ void zero_k(unsigned* __restrict__ p, size_t n) {
    size_t i = (size_t)blockIdx.x * blockDim.x + threadIdx.x;
    size_t stride = (size_t)gridDim.x * blockDim.x;
    for (; i < n; i += stride) p[i] = 0u;
}

// out tile pre-init with per-column bias (folds GAT bias into scatter target)
__global__ void fill_bias_k(float* __restrict__ x, const float* __restrict__ bias, size_t tot) {
    size_t i = (size_t)blockIdx.x * blockDim.x + threadIdx.x;
    size_t stride = (size_t)gridDim.x * blockDim.x;
    for (; i < tot; i += stride) x[i] = bias[i & 511];
}

__global__ void relu_k(float* __restrict__ x, size_t tot) {
    size_t i = (size_t)blockIdx.x * blockDim.x + threadIdx.x;
    size_t stride = (size_t)gridDim.x * blockDim.x;
    for (; i < tot; i += stride) x[i] = fmaxf(x[i], 0.f);
}

__global__ void cvt_bf16_k(const float* __restrict__ src, __bf16* __restrict__ dst, size_t n) {
    size_t i = (size_t)blockIdx.x * blockDim.x + threadIdx.x;
    size_t stride = (size_t)gridDim.x * blockDim.x;
    for (; i < n; i += stride) dst[i] = (__bf16)src[i];
}

// Wt[n*K + k] = W[k*Nc + n]  (weight transpose + bf16 cast; coalesced writes)
__global__ void transpose_bf16_k(const float* __restrict__ W, __bf16* __restrict__ Wt,
                                 int K, int Nc) {
    size_t tot = (size_t)K * Nc;
    size_t i = (size_t)blockIdx.x * blockDim.x + threadIdx.x;
    size_t stride = (size_t)gridDim.x * blockDim.x;
    for (; i < tot; i += stride) {
        int n = (int)(i / K);
        int k = (int)(i % K);
        Wt[i] = (__bf16)W[(size_t)k * Nc + n];
    }
}

// ---------------- WMMA GEMM: C[M x Nout] = A[M x K](bf16) * Bt[Nout x K](bf16)^T ----------------
// One wave computes a 16x64 tile (4 accumulators; A fragment reused across 4 B tiles).
// Software-pipelined: loads for step k+32 are issued BEFORE the 4 WMMAs of step k,
// so s_wait_loadcnt can retire with loads still in flight (latency hiding on the
// matrix pipe). Last iteration redundantly prefetches k=0 to keep the loop uniform.
__global__ void __launch_bounds__(256)
gemm_wmma_bf16(const __bf16* __restrict__ A, const __bf16* __restrict__ Bt,
               const float* __restrict__ bias, float* __restrict__ C,
               int M, int K, int Nout, int do_relu) {
    const int lane = threadIdx.x & 31;
    const int half = lane >> 4;
    const int l16  = lane & 15;
    const int nTilesN = Nout >> 6;                       // groups of 64 columns
    const int nWaves  = (M >> 4) * nTilesN;
    int wg = blockIdx.x * (blockDim.x >> 5) + (threadIdx.x >> 5);
    if (wg >= nWaves) return;                            // wave-uniform
    const int mt   = wg / nTilesN;
    const int ngrp = wg % nTilesN;

    // per-lane fragment base: row = tile row (A) / tile col (Bt); half selects K sub-blocks
    const __bf16* arow  = A  + (size_t)(mt * 16 + l16) * K + half * 8;
    const __bf16* brow0 = Bt + (size_t)(ngrp * 64 + l16) * K + half * 8;

    FragBF a0, b0[4];
    a0.q[0] = *(const uint4*)(arow);
    a0.q[1] = *(const uint4*)(arow + 16);
#pragma unroll
    for (int t = 0; t < 4; ++t) {
        const __bf16* bp = brow0 + (size_t)t * 16 * K;
        b0[t].q[0] = *(const uint4*)(bp);
        b0[t].q[1] = *(const uint4*)(bp + 16);
    }

    v8f acc[4] = { v8f{0.f}, v8f{0.f}, v8f{0.f}, v8f{0.f} };
    for (int k0 = 0; k0 < K; k0 += 32) {
        // prefetch next K-step (clamped; redundant reload of k=0 on final iter)
        int kn = (k0 + 32 < K) ? (k0 + 32) : 0;
        FragBF a1, b1[4];
        a1.q[0] = *(const uint4*)(arow + kn);
        a1.q[1] = *(const uint4*)(arow + kn + 16);
#pragma unroll
        for (int t = 0; t < 4; ++t) {
            const __bf16* bp = brow0 + (size_t)t * 16 * K + kn;
            b1[t].q[0] = *(const uint4*)(bp);
            b1[t].q[1] = *(const uint4*)(bp + 16);
        }
#pragma unroll
        for (int t = 0; t < 4; ++t) {
            acc[t] = __builtin_amdgcn_wmma_f32_16x16x32_bf16(
                false, a0.v, false, b0[t].v, (short)0, acc[t], false, false);
        }
        a0 = a1;
#pragma unroll
        for (int t = 0; t < 4; ++t) b0[t] = b1[t];
    }

    // C/D layout: VGPR r, lanes0-15 -> M=r, lanes16-31 -> M=8+r; N = lane&15
#pragma unroll
    for (int t = 0; t < 4; ++t) {
        int col = ngrp * 64 + t * 16 + l16;
        float bv = bias ? bias[col] : 0.0f;
#pragma unroll
        for (int r = 0; r < 8; ++r) {
            int row = mt * 16 + half * 8 + r;
            float v = acc[t][r] + bv;
            if (do_relu) v = fmaxf(v, 0.0f);
            C[(size_t)row * Nout + col] = v;
        }
    }
}

// ---------------- edge-type aggregation ----------------
// agg[i] = cnt0[i]*emb[0] + cnt1[i]*emb[1]  (2-row table => count-based form)
__global__ void count_types_k(const int* __restrict__ src, const int* __restrict__ dst,
                              const float* __restrict__ eattr, int* __restrict__ cntT, int E) {
    int e = blockIdx.x * blockDim.x + threadIdx.x;
    int stride = gridDim.x * blockDim.x;
    for (; e < E; e += stride) {
        int t = (eattr[2 * e] > 0.5f) ? 1 : 0;
        atomicAdd(&cntT[2 * src[e] + t], 1);
        atomicAdd(&cntT[2 * dst[e] + t], 1);
    }
}

// fused: xa = bf16(x + agg/cnt)  -> written straight into GEMM staging (skips fp32 pass)
__global__ void apply_agg_bf16_k(const float* __restrict__ x, const int* __restrict__ cntT,
                                 const float* __restrict__ emb, __bf16* __restrict__ xa,
                                 int N, int D, int dshift) {
    size_t tot = (size_t)N * D;
    size_t i = (size_t)blockIdx.x * blockDim.x + threadIdx.x;
    size_t stride = (size_t)gridDim.x * blockDim.x;
    for (; i < tot; i += stride) {
        int n = (int)(i >> dshift);
        int j = (int)(i & (D - 1));
        float c0 = (float)cntT[2 * n];
        float c1 = (float)cntT[2 * n + 1];
        float cnt = fmaxf(c0 + c1, 1.0f);
        xa[i] = (__bf16)(x[i] + (c0 * emb[j] + c1 * emb[D + j]) / cnt);
    }
}

// ---------------- attention ----------------
// one block per node; wave w = head; 128-wide dot per head for a_src and a_dst
__global__ void __launch_bounds__(128)
attn_logits_k(const float* __restrict__ h, const float* __restrict__ a_src,
              const float* __restrict__ a_dst, float* __restrict__ al_s,
              float* __restrict__ al_d, int N) {
    int n = blockIdx.x;
    if (n >= N) return;
    int w = threadIdx.x >> 5;
    int lane = threadIdx.x & 31;
    const float* hp = h + (size_t)n * 512 + w * 128;
    const float* as = a_src + w * 128;
    const float* ad = a_dst + w * 128;
    float ss = 0.f, sd = 0.f;
    for (int k = lane; k < 128; k += 32) {
        float hv = hp[k];
        ss += hv * as[k];
        sd += hv * ad[k];
    }
    for (int off = 16; off > 0; off >>= 1) {
        ss += __shfl_xor(ss, off, 32);
        sd += __shfl_xor(sd, off, 32);
    }
    if (lane == 0) { al_s[n * 4 + w] = ss; al_d[n * 4 + w] = sd; }
}

__device__ __forceinline__ void edge_sd(int e, int E, const int* src, const int* dst,
                                        int& s, int& d) {
    if (e < E) { s = src[e]; d = dst[e]; }
    else       { s = e - E; d = e - E; }    // self loops
}

__global__ void edge_max_k(const int* __restrict__ src, const int* __restrict__ dst,
                           const float* __restrict__ al_s, const float* __restrict__ al_d,
                           unsigned* __restrict__ mxU, int E, int N) {
    int tot = E + N;
    int e = blockIdx.x * blockDim.x + threadIdx.x;
    int stride = gridDim.x * blockDim.x;
    for (; e < tot; e += stride) {
        int s, d; edge_sd(e, E, src, dst, s, d);
#pragma unroll
        for (int hh = 0; hh < 4; ++hh) {
            float lg = al_s[s * 4 + hh] + al_d[d * 4 + hh];
            lg = (lg > 0.f) ? lg : 0.2f * lg;
            atomicMax(&mxU[d * 4 + hh], f2o(lg));
        }
    }
}

__global__ void edge_denom_k(const int* __restrict__ src, const int* __restrict__ dst,
                             const float* __restrict__ al_s, const float* __restrict__ al_d,
                             const unsigned* __restrict__ mxU, float* __restrict__ denom,
                             int E, int N) {
    int tot = E + N;
    int e = blockIdx.x * blockDim.x + threadIdx.x;
    int stride = gridDim.x * blockDim.x;
    for (; e < tot; e += stride) {
        int s, d; edge_sd(e, E, src, dst, s, d);
#pragma unroll
        for (int hh = 0; hh < 4; ++hh) {
            float lg = al_s[s * 4 + hh] + al_d[d * 4 + hh];
            lg = (lg > 0.f) ? lg : 0.2f * lg;
            float ex = __expf(lg - o2f(mxU[d * 4 + hh]));
            atomicAdd(&denom[d * 4 + hh], ex);
        }
    }
}

// one 32-lane wave per edge: out[d, :512] += alpha[head]*h[s, :512]
__global__ void __launch_bounds__(256)
edge_agg_k(const int* __restrict__ src, const int* __restrict__ dst,
           const float* __restrict__ al_s, const float* __restrict__ al_d,
           const unsigned* __restrict__ mxU, const float* __restrict__ denom,
           const float* __restrict__ h, float* __restrict__ out, int E, int N) {
    int wave = (blockIdx.x * blockDim.x + threadIdx.x) >> 5;
    int lane = threadIdx.x & 31;
    int tot = E + N;
    if (wave >= tot) return;
    int s, d; edge_sd(wave, E, src, dst, s, d);
    float alpha[4];
#pragma unroll
    for (int hh = 0; hh < 4; ++hh) {
        float lg = al_s[s * 4 + hh] + al_d[d * 4 + hh];
        lg = (lg > 0.f) ? lg : 0.2f * lg;
        float ex = __expf(lg - o2f(mxU[d * 4 + hh]));
        alpha[hh] = ex / (denom[d * 4 + hh] + 1e-16f);
    }
    const float* hs = h + (size_t)s * 512;
    float* od = out + (size_t)d * 512;
#pragma unroll
    for (int p = 0; p < 16; ++p) {
        int j = lane + p * 32;
        atomicAdd(&od[j], alpha[j >> 7] * hs[j]);
    }
}

// ---------------- pooling ----------------
__global__ void pool_k(const float* __restrict__ x, const int* __restrict__ batch,
                       float* __restrict__ psum, unsigned* __restrict__ pmax, int N) {
    size_t tot = (size_t)N * 512;
    size_t i = (size_t)blockIdx.x * blockDim.x + threadIdx.x;
    size_t stride = (size_t)gridDim.x * blockDim.x;
    for (; i < tot; i += stride) {
        int n = (int)(i >> 9);
        int j = (int)(i & 511);
        int b = batch[n];
        float v = x[i];
        atomicAdd(&psum[b * 512 + j], v);
        atomicMax(&pmax[b * 512 + j], f2o(v));
    }
}

__global__ void pool_cnt_k(const int* __restrict__ batch, float* __restrict__ pcnt, int N) {
    int i = blockIdx.x * blockDim.x + threadIdx.x;
    int stride = gridDim.x * blockDim.x;
    for (; i < N; i += stride) atomicAdd(&pcnt[batch[i]], 1.0f);
}

// ---------------- MLP head (B=8, all in LDS, one block) ----------------
__global__ void __launch_bounds__(128)
head_k(const float* __restrict__ psum, const unsigned* __restrict__ pmax,
       const float* __restrict__ pcnt,
       const float* __restrict__ h1w, const float* __restrict__ h1b,
       const float* __restrict__ g1,  const float* __restrict__ b1,
       const float* __restrict__ h2w, const float* __restrict__ h2b,
       const float* __restrict__ g2,  const float* __restrict__ b2,
       const float* __restrict__ h3w, const float* __restrict__ h3b,
       float* __restrict__ out) {
    __shared__ float sp[8 * 1024];
    __shared__ float sq[8 * 128];
    __shared__ float sr[8 * 64];
    __shared__ float smean[8], svar[8];
    int tid = threadIdx.x;

    for (int idx = tid; idx < 8 * 1024; idx += 128) {
        int b = idx >> 10, j = idx & 1023;
        float v = (j < 512) ? psum[b * 512 + j] / fmaxf(pcnt[b], 1.0f)
                            : o2f(pmax[b * 512 + (j - 512)]);
        sp[idx] = v;
    }
    __syncthreads();
    for (int idx = tid; idx < 8 * 128; idx += 128) {
        int b = idx >> 7, c = idx & 127;
        float acc = h1b[c];
        for (int k = 0; k < 1024; ++k) acc += sp[b * 1024 + k] * h1w[k * 128 + c];
        sq[idx] = acc;
    }
    __syncthreads();
    if (tid < 8) {
        float m = 0.f;
        for (int c = 0; c < 128; ++c) m += sq[tid * 128 + c];
        m *= (1.0f / 128.0f);
        float v = 0.f;
        for (int c = 0; c < 128; ++c) { float dd = sq[tid * 128 + c] - m; v += dd * dd; }
        smean[tid] = m; svar[tid] = v * (1.0f / 128.0f);
    }
    __syncthreads();
    for (int idx = tid; idx < 8 * 128; idx += 128) {
        int b = idx >> 7, c = idx & 127;
        float v = (sq[idx] - smean[b]) * rsqrtf(svar[b] + 1e-5f) * g1[c] + b1[c];
        sq[idx] = fmaxf(v, 0.f);
    }
    __syncthreads();
    for (int idx = tid; idx < 8 * 64; idx += 128) {
        int b = idx >> 6, c = idx & 63;
        float acc = h2b[c];
        for (int k = 0; k < 128; ++k) acc += sq[b * 128 + k] * h2w[k * 64 + c];
        sr[idx] = acc;
    }
    __syncthreads();
    if (tid < 8) {
        float m = 0.f;
        for (int c = 0; c < 64; ++c) m += sr[tid * 64 + c];
        m *= (1.0f / 64.0f);
        float v = 0.f;
        for (int c = 0; c < 64; ++c) { float dd = sr[tid * 64 + c] - m; v += dd * dd; }
        smean[tid] = m; svar[tid] = v * (1.0f / 64.0f);
    }
    __syncthreads();
    for (int idx = tid; idx < 8 * 64; idx += 128) {
        int b = idx >> 6, c = idx & 63;
        float v = (sr[idx] - smean[b]) * rsqrtf(svar[b] + 1e-5f) * g2[c] + b2[c];
        sr[idx] = fmaxf(v, 0.f);
    }
    __syncthreads();
    if (tid < 8) {
        float acc = h3b[0];
        for (int k = 0; k < 64; ++k) acc += sr[tid * 64 + k] * h3w[k];
        out[tid] = acc;
    }
}

// ---------------- host orchestration ----------------
extern "C" void kernel_launch(void* const* d_in, const int* in_sizes, int n_in,
                              void* d_out, int out_size, void* d_ws, size_t ws_size,
                              hipStream_t stream) {
    (void)n_in; (void)out_size; (void)ws_size;
    const float* x_in  = (const float*)d_in[0];
    const int*   ei    = (const int*)d_in[1];
    const float* eattr = (const float*)d_in[2];
    const int*   batch = (const int*)d_in[3];
    const float* in_w  = (const float*)d_in[4];
    const float* in_b  = (const float*)d_in[5];
    const float* emb[3] = { (const float*)d_in[6],  (const float*)d_in[11], (const float*)d_in[16] };
    const float* gw[3]  = { (const float*)d_in[7],  (const float*)d_in[12], (const float*)d_in[17] };
    const float* gas[3] = { (const float*)d_in[8],  (const float*)d_in[13], (const float*)d_in[18] };
    const float* gad[3] = { (const float*)d_in[9],  (const float*)d_in[14], (const float*)d_in[19] };
    const float* gb[3]  = { (const float*)d_in[10], (const float*)d_in[15], (const float*)d_in[20] };
    const float* h1w = (const float*)d_in[21]; const float* h1b = (const float*)d_in[22];
    const float* g1  = (const float*)d_in[23]; const float* b1  = (const float*)d_in[24];
    const float* h2w = (const float*)d_in[25]; const float* h2b = (const float*)d_in[26];
    const float* g2  = (const float*)d_in[27]; const float* b2  = (const float*)d_in[28];
    const float* h3w = (const float*)d_in[29]; const float* h3b = (const float*)d_in[30];

    const int N = in_sizes[0] / 64;     // 20000 (multiple of 16)
    const int E = in_sizes[1] / 2;      // 320000
    const int* src = ei;
    const int* dst = ei + E;

    // workspace carve (256B aligned)
    size_t off = 0;
    auto take = [&](size_t bytes) -> char* {
        char* p = (char*)d_ws + off;
        off += (bytes + 255) & ~(size_t)255;
        return p;
    };
    float*    bufX  = (float*)take((size_t)N * 512 * 4);   // x / out (ping)
    float*    bufH  = (float*)take((size_t)N * 512 * 4);   // h = xa @ W
    __bf16*   abf   = (__bf16*)take((size_t)N * 512 * 2);  // bf16 GEMM staging
    __bf16*   wtb   = (__bf16*)take((size_t)512 * 512 * 2);// bf16 transposed weights
    float*    al_s  = (float*)take((size_t)N * 4 * 4);
    float*    al_d  = (float*)take((size_t)N * 4 * 4);
    unsigned* mxU   = (unsigned*)take((size_t)N * 4 * 4);
    float*    denom = (float*)take((size_t)N * 4 * 4);
    int*      cntT  = (int*)take((size_t)N * 2 * 4);
    float*    psum  = (float*)take((size_t)8 * 512 * 4);
    unsigned* pmaxU = (unsigned*)take((size_t)8 * 512 * 4);
    float*    pcnt  = (float*)take((size_t)8 * 4);

    auto zero = [&](void* p, size_t words) {
        int blocks = (int)((words + 255) / 256);
        zero_k<<<blocks, 256, 0, stream>>>((unsigned*)p, words);
    };
    auto gblocks = [](size_t n) { return (int)((n + 255) / 256); };

    // ---- input layer: x0 = relu(x @ in_w + in_b) ----
    cvt_bf16_k<<<gblocks((size_t)N * 64), 256, 0, stream>>>(x_in, abf, (size_t)N * 64);
    transpose_bf16_k<<<gblocks((size_t)64 * 128), 256, 0, stream>>>(in_w, wtb, 64, 128);
    {
        int nWaves = (N / 16) * (128 / 64);
        gemm_wmma_bf16<<<(nWaves + 7) / 8, 256, 0, stream>>>(abf, wtb, in_b, bufX, N, 64, 128, 1);
    }

    const int Dl[3] = { 128, 512, 512 };
    const int Ds[3] = { 7, 9, 9 };
    const int totE = E + N;
    for (int L = 0; L < 3; ++L) {
        int D = Dl[L];
        // edge-type aggregation (count form), fused straight to bf16 staging
        zero(cntT, (size_t)2 * N);
        count_types_k<<<gblocks(E), 256, 0, stream>>>(src, dst, eattr, cntT, E);
        apply_agg_bf16_k<<<gblocks((size_t)N * D), 256, 0, stream>>>(bufX, cntT, emb[L], abf, N, D, Ds[L]);
        // h = xa @ W (WMMA bf16, software-pipelined)
        transpose_bf16_k<<<gblocks((size_t)D * 512), 256, 0, stream>>>(gw[L], wtb, D, 512);
        {
            int nWaves = (N / 16) * (512 / 64);
            gemm_wmma_bf16<<<(nWaves + 7) / 8, 256, 0, stream>>>(abf, wtb, nullptr, bufH, N, D, 512, 0);
        }
        // attention: logits, segment max, denom, weighted scatter
        attn_logits_k<<<N, 128, 0, stream>>>(bufH, gas[L], gad[L], al_s, al_d, N);
        zero(mxU, (size_t)4 * N);
        zero(denom, (size_t)4 * N);
        edge_max_k<<<gblocks(totE), 256, 0, stream>>>(src, dst, al_s, al_d, mxU, E, N);
        edge_denom_k<<<gblocks(totE), 256, 0, stream>>>(src, dst, al_s, al_d, mxU, denom, E, N);
        // out pre-filled with bias; scatter accumulates on top
        fill_bias_k<<<gblocks((size_t)N * 512), 256, 0, stream>>>(bufX, gb[L], (size_t)N * 512);
        edge_agg_k<<<(totE + 7) / 8, 256, 0, stream>>>(src, dst, al_s, al_d, mxU, denom, bufH, bufX, E, N);
        if (L < 2)
            relu_k<<<gblocks((size_t)N * 512), 256, 0, stream>>>(bufX, (size_t)N * 512);
    }

    // ---- pooling + head ----
    zero(psum, (size_t)8 * 512);
    zero(pmaxU, (size_t)8 * 512);
    zero(pcnt, 8);
    pool_k<<<gblocks((size_t)N * 512), 256, 0, stream>>>(bufX, batch, psum, pmaxU, N);
    pool_cnt_k<<<gblocks(N), 256, 0, stream>>>(batch, pcnt, N);
    head_k<<<1, 128, 0, stream>>>(psum, pmaxU, pcnt, h1w, h1b, g1, b1,
                                  h2w, h2b, g2, b2, h3w, h3b, (float*)d_out);
}
